// RNN_74431783239825
// MI455X (gfx1250) — matrix operand order repaired and checked by reading, not compiled, not run on hardware
//
#include <hip/hip_runtime.h>
#include <hip/hip_bf16.h>
#include <math.h>

typedef _Float16 h16;
typedef __attribute__((ext_vector_type(16))) _Float16 v16h;
typedef __attribute__((ext_vector_type(8)))  _Float16 v8h;
typedef __attribute__((ext_vector_type(4)))  _Float16 v4h;
typedef __attribute__((ext_vector_type(8)))  float    v8f;
typedef __attribute__((ext_vector_type(4)))  float    v4f;

#define B_    64
#define T_    512
#define NIN_  64
#define N_    1024
#define NOUT_ 32
#define NWG_  32
#define NC_   32      // hidden columns per workgroup
#define ALPHA_ 0.1f

// ---------------------------------------------------------------------------
// init: zero the barrier counter, fill r_0 = retanh(x_0) broadcast over batch
// ---------------------------------------------------------------------------
__global__ __launch_bounds__(256) void rnn_init_kernel(const float* __restrict__ x0,
                                                       h16* __restrict__ rbuf0,
                                                       unsigned* __restrict__ bar) {
  int idx = blockIdx.x * 256 + threadIdx.x;
  if (idx == 0) *bar = 0u;
  if (idx < B_ * N_) {
    float v = fmaxf(tanhf(x0[idx & (N_ - 1)]), 0.0f);
    rbuf0[idx] = (h16)v;
  }
}

// ---------------------------------------------------------------------------
// persistent RNN kernel: 32 WGs x 256 threads, one grid barrier per timestep
// ---------------------------------------------------------------------------
__global__ __launch_bounds__(256, 1) void rnn_main_kernel(
    const float* __restrict__ u,    const float* __restrict__ sn,
    const float* __restrict__ rn,   const float* __restrict__ on,
    const float* __restrict__ Win,  const float* __restrict__ bin,
    const float* __restrict__ Wrec, const float* __restrict__ Wout,
    const float* __restrict__ bout,
    float* __restrict__ X, float* __restrict__ R, float* __restrict__ Z,
    h16* __restrict__ rbuf0, h16* __restrict__ rbuf1,
    unsigned* __restrict__ bar, const float* __restrict__ x0) {
  __shared__ h16 WrecL[NC_ * N_];   // 64 KB: this WG's 32 rows of W_rec, fp16

  const int g    = blockIdx.x;
  const int tid  = threadIdx.x;
  const int lane = tid & 31;
  const int wave = tid >> 5;

  // ---- stage W_rec slice into LDS (f32 -> f16), reused for all 512 steps ----
  for (int i = tid * 4; i < NC_ * N_; i += 256 * 4) {
    const int nl = i >> 10;
    const int k  = i & (N_ - 1);
    v4f w = *(const v4f*)(Wrec + (size_t)(g * NC_ + nl) * N_ + k);
    v4h h; h[0] = (h16)w[0]; h[1] = (h16)w[1]; h[2] = (h16)w[2]; h[3] = (h16)w[3];
    *(v4h*)(WrecL + i) = h;
  }
  __syncthreads();

  // ---- per-wave WMMA tile coordinates (wave32, 16x16x32 f16) ----
  const int mt   = wave & 3;            // batch tile: rows mt*16..mt*16+15
  const int nt   = wave >> 2;           // hidden tile within WG slice (0/1)
  const int lm   = lane & 15;
  const int hi   = lane >> 4;           // half-wave selector
  const int m_a  = mt * 16 + lm;        // A-matrix row (batch) for this lane
  const int nloc = nt * 16 + lm;        // B-matrix column (local hidden)
  const int n_glob = g * NC_ + nloc;    // global hidden index for this lane
  const int aoff = hi * 8;              // A K-offset per ISA 16-bit A layout
  const int boff = hi * 8;              // B K-offset per ISA 16-bit B layout

  const float binv = bin[n_glob];
  const float x0v  = x0[n_glob];

  // W_in B tiles (K=64 -> 2 chunks) kept in registers for the whole loop
  v16h buin[2];
  #pragma unroll
  for (int kk = 0; kk < 2; ++kk) {
    const float* wr = Win + (size_t)n_glob * NIN_ + kk * 32 + boff;
    #pragma unroll
    for (int j = 0; j < 16; ++j) buin[kk][j] = (h16)wr[(j & 7) + ((j >> 3) << 4)];
  }

  // x carry in WMMA C layout: VGPR j -> (m = mt*16 + 8*hi + j, n = n_glob)
  v8f x;
  #pragma unroll
  for (int j = 0; j < 8; ++j) x[j] = x0v;

  #pragma unroll 1
  for (int t = 0; t < T_; ++t) {
    const h16* __restrict__ cur = (t & 1) ? rbuf1 : rbuf0;
    h16* __restrict__ nxt       = (t & 1) ? rbuf0 : rbuf1;

    v8f c = {};

    // recurrent GEMM: c += r_prev[64x1024] x W_rec_slice^T   (K = 1024)
    const h16* __restrict__ arow = cur + (size_t)m_a * N_;
    const h16* __restrict__ brow = WrecL + nloc * N_;
    #pragma unroll 4
    for (int k0 = 0; k0 < N_; k0 += 32) {
      v8h a0 = *(const v8h*)(arow + k0 + aoff);
      v8h a1 = *(const v8h*)(arow + k0 + 16 + aoff);
      v16h a = __builtin_shufflevector(a0, a1,
                 0,1,2,3,4,5,6,7,8,9,10,11,12,13,14,15);
      v8h b0 = *(const v8h*)(brow + k0 + boff);
      v8h b1 = *(const v8h*)(brow + k0 + 16 + boff);
      v16h b = __builtin_shufflevector(b0, b1,
                 0,1,2,3,4,5,6,7,8,9,10,11,12,13,14,15);          // LDS, ds_load_b128 x2
      c = __builtin_amdgcn_wmma_f32_16x16x32_f16(false, a, false, b,
                                                 (short)0, c, false, false);
    }

    // input GEMM: c += u_t[64x64] x W_in_slice^T  (K = 64, cvt f32->f16 inline)
    const float* __restrict__ urow = u + (size_t)m_a * T_ * NIN_ + (size_t)t * NIN_;
    #pragma unroll
    for (int kk = 0; kk < 2; ++kk) {
      v4f f0 = *(const v4f*)(urow + kk * 32 + aoff);
      v4f f1 = *(const v4f*)(urow + kk * 32 + aoff + 4);
      v4f f2 = *(const v4f*)(urow + kk * 32 + 16 + aoff);
      v4f f3 = *(const v4f*)(urow + kk * 32 + 16 + aoff + 4);
      v16h a;
      #pragma unroll
      for (int j = 0; j < 4; ++j) {
        a[j]      = (h16)f0[j];
        a[4 + j]  = (h16)f1[j];
        a[8 + j]  = (h16)f2[j];
        a[12 + j] = (h16)f3[j];
      }
      c = __builtin_amdgcn_wmma_f32_16x16x32_f16(false, a, false, buin[kk],
                                                 (short)0, c, false, false);
    }

    // ---- elementwise state update, write X / R / r_f16(next) ----
    #pragma unroll
    for (int j = 0; j < 8; ++j) {
      const int m = mt * 16 + 8 * hi + j;                  // batch index
      const size_t off = (size_t)m * T_ * N_ + (size_t)t * N_ + n_glob;
      const float xv = (1.0f - ALPHA_) * x[j] + ALPHA_ * (c[j] + binv + sn[off]);
      X[off] = xv;
      const float rv = fmaxf(tanhf(xv), 0.0f) + rn[off];
      R[off] = rv;
      x[j] = xv;
      nxt[(size_t)m * N_ + n_glob] = (h16)rv;
    }

    // ---- grid-wide release/acquire barrier: publish r_t to all WGs ----
    __builtin_amdgcn_fence(__ATOMIC_RELEASE, "agent");
    __syncthreads();
    if (tid == 0) {
      __hip_atomic_fetch_add(bar, 1u, __ATOMIC_RELEASE, __HIP_MEMORY_SCOPE_AGENT);
      const unsigned tgt = (unsigned)(t + 1) * (unsigned)NWG_;
      while (__hip_atomic_load(bar, __ATOMIC_RELAXED, __HIP_MEMORY_SCOPE_AGENT) < tgt) {
        __builtin_amdgcn_s_sleep(2);
      }
    }
    __syncthreads();
    __builtin_amdgcn_fence(__ATOMIC_ACQUIRE, "agent");

    // ---- readout z_t for this WG's 2 batch rows from the full r_t ----
    {
      const int p     = tid >> 2;
      const int sub   = tid & 3;
      const int brow_ = 2 * g + (p >> 5);
      const int o     = p & 31;
      const h16*   rrow = nxt  + (size_t)brow_ * N_ + sub * 256;
      const float* wrow = Wout + (size_t)o * N_ + sub * 256;
      float acc = 0.0f;
      #pragma unroll 4
      for (int k = 0; k < 256; k += 8) {
        v8h rv = *(const v8h*)(rrow + k);
        v4f w0 = *(const v4f*)(wrow + k);
        v4f w1 = *(const v4f*)(wrow + k + 4);
        #pragma unroll
        for (int j = 0; j < 4; ++j) acc += (float)rv[j] * w0[j];
        #pragma unroll
        for (int j = 0; j < 4; ++j) acc += (float)rv[4 + j] * w1[j];
      }
      acc += __shfl_xor(acc, 1, 32);
      acc += __shfl_xor(acc, 2, 32);
      if (sub == 0) {
        const size_t zoff = (size_t)brow_ * T_ * NOUT_ + (size_t)t * NOUT_ + o;
        Z[zoff] = acc + bout[o] + on[zoff];
      }
    }
  }
}

// ---------------------------------------------------------------------------
extern "C" void kernel_launch(void* const* d_in, const int* in_sizes, int n_in,
                              void* d_out, int out_size, void* d_ws, size_t ws_size,
                              hipStream_t stream) {
  (void)in_sizes; (void)n_in; (void)out_size; (void)ws_size;
  const float* u    = (const float*)d_in[0];
  const float* x0   = (const float*)d_in[1];
  const float* sn   = (const float*)d_in[2];
  const float* rn   = (const float*)d_in[3];
  const float* on   = (const float*)d_in[4];
  const float* Win  = (const float*)d_in[5];
  const float* bin  = (const float*)d_in[6];
  const float* Wrec = (const float*)d_in[7];
  const float* Wout = (const float*)d_in[8];
  const float* bout = (const float*)d_in[9];

  float* X = (float*)d_out;
  float* R = X + (size_t)B_ * T_ * N_;
  float* Z = R + (size_t)B_ * T_ * N_;

  unsigned* bar = (unsigned*)d_ws;
  h16* rbuf0 = (h16*)((char*)d_ws + 256);
  h16* rbuf1 = rbuf0 + (size_t)B_ * N_;

  rnn_init_kernel<<<(B_ * N_ + 255) / 256, 256, 0, stream>>>(x0, rbuf0, bar);
  rnn_main_kernel<<<NWG_, 256, 0, stream>>>(u, sn, rn, on, Win, bin, Wrec, Wout,
                                            bout, X, R, Z, rbuf0, rbuf1, bar, x0);
}
